// DGNN_21182778704834
// MI455X (gfx1250) — compile-verified, matrix operand order checked by reading, not run on hardware
//
#include <hip/hip_runtime.h>
#include <hip/hip_bf16.h>
#include <math.h>

typedef __bf16 bf16_t;
typedef __attribute__((ext_vector_type(16))) __bf16 v16bf;
typedef __attribute__((ext_vector_type(4)))  __bf16 v4bf;
typedef __attribute__((ext_vector_type(8)))  float  v8f;

#define B_EV  4096
#define H_LEN 20
#define FEATD 128

// workspace byte offsets (all 256B aligned)
#define OFF_SOFT1   0u          // B*H f32      = 327680
#define OFF_W1B     327680u     // 128*128 bf16 = 32768
#define OFF_W2B     360448u
#define OFF_W3B     393216u
#define OFF_W4B     425984u
#define OFF_B12     458752u     // 128 f32
#define OFF_B34     459264u
#define OFF_AGG1    459776u     // B*128 bf16 = 1048576
#define OFF_SELFB   1508352u
#define OFF_XS1     2556928u
#define OFF_XAGG    3605504u

__device__ __forceinline__ v8f wmma_bf16(v16bf a, v16bf b, v8f c) {
    return __builtin_amdgcn_wmma_f32_16x16x32_bf16(false, a, false, b, (short)0, c, false, false);
}

// ---------------- K0: weight conversion + bias folding ----------------
__global__ void k_prep(const float* w1, const float* w2, const float* w3, const float* w4,
                       const float* b1, const float* b2, const float* b3, const float* b4,
                       bf16_t* w1b, bf16_t* w2b, bf16_t* w3b, bf16_t* w4b,
                       float* b12, float* b34)
{
    int idx = blockIdx.x * 256 + threadIdx.x;
    if (idx < 4 * FEATD * FEATD) {
        int wi = idx >> 14, j = idx & 16383;
        const float* s = (wi == 0) ? w1 : (wi == 1) ? w2 : (wi == 2) ? w3 : w4;
        bf16_t*      d = (wi == 0) ? w1b : (wi == 1) ? w2b : (wi == 2) ? w3b : w4b;
        d[j] = (bf16_t)s[j];
    } else if (idx < 4 * FEATD * FEATD + FEATD) {
        int n = idx - 4 * FEATD * FEATD;
        b12[n] = b1[n] + b2[n];
    } else if (idx < 4 * FEATD * FEATD + 2 * FEATD) {
        int n = idx - 4 * FEATD * FEATD - FEATD;
        b34[n] = b3[n] + b4[n];
    }
}

// ---------------- K1: softmax over history (per event) ----------------
__global__ void k_soft1(const float* e_time, const float* his_time, const float* delta1,
                        float* soft1)
{
    int b = blockIdx.x * 256 + threadIdx.x;
    if (b >= B_EV) return;
    float d = delta1[0], et = e_time[b];
    float v[H_LEN];
    float m = -3.4e38f;
#pragma unroll
    for (int h = 0; h < H_LEN; h++) {
        float x = -d * (et - his_time[b * H_LEN + h]);
        v[h] = x;
        m = fmaxf(m, x);
    }
    float s = 0.f;
#pragma unroll
    for (int h = 0; h < H_LEN; h++) { float e = __expf(v[h] - m); v[h] = e; s += e; }
    float inv = 1.0f / s;
#pragma unroll
    for (int h = 0; h < H_LEN; h++) soft1[b * H_LEN + h] = v[h] * inv;
}

// ---------------- K2: agg1 = soft1-weighted one_hop sum (bf16); self_feat -> bf16 ----------------
__global__ void k_agg1_selfcvt(const float* one_hop, const float* self_feat,
                               const float* soft1, bf16_t* agg1b, bf16_t* selfb)
{
    int gw   = (blockIdx.x * blockDim.x + threadIdx.x) >> 5;  // global wave id
    int lane = threadIdx.x & 31;
    if (gw < B_EV) {
        int b = gw;
        float ax = 0.f, ay = 0.f, az = 0.f, aw = 0.f;
        for (int h = 0; h < H_LEN; h++) {
            float wg = soft1[b * H_LEN + h];
            const float4 f = reinterpret_cast<const float4*>(one_hop)[(size_t)(b * H_LEN + h) * 32 + lane];
            ax += wg * f.x; ay += wg * f.y; az += wg * f.z; aw += wg * f.w;
        }
        v4bf o; o[0] = (bf16_t)ax; o[1] = (bf16_t)ay; o[2] = (bf16_t)az; o[3] = (bf16_t)aw;
        *reinterpret_cast<v4bf*>(agg1b + (size_t)b * FEATD + lane * 4) = o;
    } else if (gw < 2 * B_EV) {
        int b = gw - B_EV;
        const float4 f = reinterpret_cast<const float4*>(self_feat)[(size_t)b * 32 + lane];
        v4bf o; o[0] = (bf16_t)f.x; o[1] = (bf16_t)f.y; o[2] = (bf16_t)f.z; o[3] = (bf16_t)f.w;
        *reinterpret_cast<v4bf*>(selfb + (size_t)b * FEATD + lane * 4) = o;
    }
}

// ---------------- K3: fused one-hop pipeline ----------------
// block = 4 events (80 one-hop rows). Phase 1: stream two_hop once, softmax2-weighted
// aggregation into LDS (bf16); one_hop rows cvt to bf16 in LDS. Phase 2: dual WMMA GEMM
// (one_hop@w1T + agg2@w2T) + bias + relu + soft1-weighted row reduction -> xagg (bf16).
__global__ __launch_bounds__(256) void k_onehop(
    const float* two_hop, const float* one_hop,
    const float* his_time, const float* his_his,
    const float* delta1, const float* soft1,
    const bf16_t* w1b, const bf16_t* w2b, const float* b12,
    bf16_t* xaggb)
{
    __shared__ bf16_t sOne[80 * FEATD];   // 20 KB
    __shared__ bf16_t sAgg[80 * FEATD];   // 20 KB
    const int blk = blockIdx.x;           // 0..1023
    const int tid = threadIdx.x;
    const int w = tid >> 5, lane = tid & 31;
    const int l16 = lane & 15, lhalf = lane >> 4;

    // phase 1a: stage one_hop rows (f32 -> bf16) into LDS
    for (int q = tid; q < 80 * 32; q += 256) {
        int row = q >> 5, c4 = q & 31;
        const float4 f = reinterpret_cast<const float4*>(one_hop)[(size_t)(blk * 80 + row) * 32 + c4];
        v4bf o; o[0] = (bf16_t)f.x; o[1] = (bf16_t)f.y; o[2] = (bf16_t)f.z; o[3] = (bf16_t)f.w;
        *reinterpret_cast<v4bf*>(&sOne[row * FEATD + c4 * 4]) = o;
    }

    // phase 1b: softmax2-weighted two_hop aggregation (10 rows per wave)
    const float d = delta1[0];
    for (int i = 0; i < 10; i++) {
        int hr = w * 10 + i;
        int R  = blk * 80 + hr;           // == b*20 + h
        float ht  = his_time[R];
        float val = (lane < H_LEN) ? (-d * (ht - his_his[(size_t)R * H_LEN + lane])) : -3.4e38f;
        float m = val;
        for (int off = 16; off > 0; off >>= 1) m = fmaxf(m, __shfl_xor(m, off, 32));
        float e = (lane < H_LEN) ? __expf(val - m) : 0.0f;
        float s = e;
        for (int off = 16; off > 0; off >>= 1) s += __shfl_xor(s, off, 32);
        float wgt = e / s;
        float ax = 0.f, ay = 0.f, az = 0.f, aw = 0.f;
        for (int h2 = 0; h2 < H_LEN; h2++) {
            float g = __shfl(wgt, h2, 32);
            const float4 f = reinterpret_cast<const float4*>(two_hop)[((size_t)R * H_LEN + h2) * 32 + lane];
            ax += g * f.x; ay += g * f.y; az += g * f.z; aw += g * f.w;
        }
        v4bf o; o[0] = (bf16_t)ax; o[1] = (bf16_t)ay; o[2] = (bf16_t)az; o[3] = (bf16_t)aw;
        *reinterpret_cast<v4bf*>(&sAgg[hr * FEATD + lane * 4]) = o;
    }
    __syncthreads();

    // phase 2: wave w owns output columns [16w, 16w+16)
    const int n = w * 16 + l16;
    v16bf wb1[4], wb2[4];
#pragma unroll
    for (int ks = 0; ks < 4; ks++) {
        wb1[ks] = *reinterpret_cast<const v16bf*>(w1b + (size_t)n * FEATD + ks * 32 + lhalf * 16);
        wb2[ks] = *reinterpret_cast<const v16bf*>(w2b + (size_t)n * FEATD + ks * 32 + lhalf * 16);
    }
    const float bias = b12[n];
    float xl0 = 0.f, xl1 = 0.f, xl2 = 0.f, xl3 = 0.f;
    for (int rt = 0; rt < 5; rt++) {
        v8f acc = {0.f, 0.f, 0.f, 0.f, 0.f, 0.f, 0.f, 0.f};
#pragma unroll
        for (int ks = 0; ks < 4; ks++) {
            v16bf a1 = *reinterpret_cast<const v16bf*>(&sOne[(rt * 16 + l16) * FEATD + ks * 32 + lhalf * 16]);
            v16bf a2 = *reinterpret_cast<const v16bf*>(&sAgg[(rt * 16 + l16) * FEATD + ks * 32 + lhalf * 16]);
            acc = wmma_bf16(a1, wb1[ks], acc);
            acc = wmma_bf16(a2, wb2[ks], acc);
        }
#pragma unroll
        for (int r = 0; r < 8; r++) {
            int lr = rt * 16 + r + 8 * lhalf;          // local row 0..79
            float v = fmaxf(acc[r] + bias, 0.0f);
            float t = soft1[blk * 80 + lr] * v;
            int lb = lr / 20;
            if      (lb == 0) xl0 += t;
            else if (lb == 1) xl1 += t;
            else if (lb == 2) xl2 += t;
            else              xl3 += t;
        }
    }
    // combine the two half-wave row groups (same column n)
    xl0 += __shfl_xor(xl0, 16, 32);
    xl1 += __shfl_xor(xl1, 16, 32);
    xl2 += __shfl_xor(xl2, 16, 32);
    xl3 += __shfl_xor(xl3, 16, 32);
    if (lane < 16) {
        int col = w * 16 + lane;
        xaggb[(size_t)(blk * 4 + 0) * FEATD + col] = (bf16_t)xl0;
        xaggb[(size_t)(blk * 4 + 1) * FEATD + col] = (bf16_t)xl1;
        xaggb[(size_t)(blk * 4 + 2) * FEATD + col] = (bf16_t)xl2;
        xaggb[(size_t)(blk * 4 + 3) * FEATD + col] = (bf16_t)xl3;
    }
}

// ---------------- K4/K5: dual-GEMM  out = A1@W1T + A2@W2T + bias ----------------
// relu_bf16out=1: relu + bf16 store (self path); 0: f32 store (final output)
__global__ __launch_bounds__(256) void k_dualgemm(
    const bf16_t* A1, const bf16_t* A2,
    const bf16_t* W1, const bf16_t* W2, const float* biasv,
    bf16_t* outb, float* outf, int relu_bf16out)
{
    const int rb = blockIdx.x;                  // row tile: rows [16rb, 16rb+16)
    const int w = threadIdx.x >> 5, lane = threadIdx.x & 31;
    const int l16 = lane & 15, lhalf = lane >> 4;
    const int n = w * 16 + l16;
    v16bf wb1[4], wb2[4];
#pragma unroll
    for (int ks = 0; ks < 4; ks++) {
        wb1[ks] = *reinterpret_cast<const v16bf*>(W1 + (size_t)n * FEATD + ks * 32 + lhalf * 16);
        wb2[ks] = *reinterpret_cast<const v16bf*>(W2 + (size_t)n * FEATD + ks * 32 + lhalf * 16);
    }
    v8f acc = {0.f, 0.f, 0.f, 0.f, 0.f, 0.f, 0.f, 0.f};
#pragma unroll
    for (int ks = 0; ks < 4; ks++) {
        v16bf a1 = *reinterpret_cast<const v16bf*>(A1 + (size_t)(rb * 16 + l16) * FEATD + ks * 32 + lhalf * 16);
        v16bf a2 = *reinterpret_cast<const v16bf*>(A2 + (size_t)(rb * 16 + l16) * FEATD + ks * 32 + lhalf * 16);
        acc = wmma_bf16(a1, wb1[ks], acc);
        acc = wmma_bf16(a2, wb2[ks], acc);
    }
    const float bias = biasv[n];
    if (relu_bf16out) {
#pragma unroll
        for (int r = 0; r < 8; r++) {
            int m = rb * 16 + r + 8 * lhalf;
            outb[(size_t)m * FEATD + n] = (bf16_t)fmaxf(acc[r] + bias, 0.0f);
        }
    } else {
#pragma unroll
        for (int r = 0; r < 8; r++) {
            int m = rb * 16 + r + 8 * lhalf;
            outf[(size_t)m * FEATD + n] = acc[r] + bias;
        }
    }
}

extern "C" void kernel_launch(void* const* d_in, const int* in_sizes, int n_in,
                              void* d_out, int out_size, void* d_ws, size_t ws_size,
                              hipStream_t stream) {
    const float* self_feat = (const float*)d_in[0];
    const float* one_hop   = (const float*)d_in[1];
    const float* two_hop   = (const float*)d_in[2];
    const float* e_time    = (const float*)d_in[3];
    const float* his_time  = (const float*)d_in[4];
    const float* his_his   = (const float*)d_in[5];
    const float* w1 = (const float*)d_in[6];  const float* b1 = (const float*)d_in[7];
    const float* w2 = (const float*)d_in[8];  const float* b2 = (const float*)d_in[9];
    const float* w3 = (const float*)d_in[10]; const float* b3 = (const float*)d_in[11];
    const float* w4 = (const float*)d_in[12]; const float* b4 = (const float*)d_in[13];
    const float* delta1 = (const float*)d_in[14];

    char* ws = (char*)d_ws;
    float*  soft1 = (float*)(ws + OFF_SOFT1);
    bf16_t* w1b   = (bf16_t*)(ws + OFF_W1B);
    bf16_t* w2b   = (bf16_t*)(ws + OFF_W2B);
    bf16_t* w3b   = (bf16_t*)(ws + OFF_W3B);
    bf16_t* w4b   = (bf16_t*)(ws + OFF_W4B);
    float*  b12   = (float*)(ws + OFF_B12);
    float*  b34   = (float*)(ws + OFF_B34);
    bf16_t* agg1b = (bf16_t*)(ws + OFF_AGG1);
    bf16_t* selfb = (bf16_t*)(ws + OFF_SELFB);
    bf16_t* xs1b  = (bf16_t*)(ws + OFF_XS1);
    bf16_t* xaggb = (bf16_t*)(ws + OFF_XAGG);
    float*  out   = (float*)d_out;

    k_prep<<<257, 256, 0, stream>>>(w1, w2, w3, w4, b1, b2, b3, b4,
                                    w1b, w2b, w3b, w4b, b12, b34);
    k_soft1<<<16, 256, 0, stream>>>(e_time, his_time, delta1, soft1);
    k_agg1_selfcvt<<<1024, 256, 0, stream>>>(one_hop, self_feat, soft1, agg1b, selfb);
    k_onehop<<<1024, 256, 0, stream>>>(two_hop, one_hop, his_time, his_his, delta1,
                                       soft1, w1b, w2b, b12, xaggb);
    // self path: x_s_one = relu(self@w1T + agg1@w2T + b1 + b2)  (bf16)
    k_dualgemm<<<256, 256, 0, stream>>>(selfb, agg1b, w1b, w2b, b12, xs1b, nullptr, 1);
    // final: out = x_s_one@w3T + xagg@w4T + b3 + b4  (f32)
    k_dualgemm<<<256, 256, 0, stream>>>(xs1b, xaggb, w3b, w4b, b34, nullptr, out, 0);
}
